// TheoRecurrentExecutor_70961449664652
// MI455X (gfx1250) — compile-verified, matrix operand order checked by reading, not run on hardware
//
#include <hip/hip_runtime.h>
#include <hip/hip_bf16.h>

typedef __attribute__((ext_vector_type(16))) _Float16 v16h;
typedef __attribute__((ext_vector_type(8)))  _Float16 v8h;
typedef __attribute__((ext_vector_type(8)))  float    v8f;

#define HID        256
#define IN_DIM     192
#define SPIKE_DIM  128
#define COORD_DIM  64
#define NUM_STEPS  25
#define BM         32          // batch rows per block
#define XPAD       200         // padded f16 stride for x tile in LDS
#define SPAD       264         // padded f16 stride for spike tile in LDS

// ---------------------------------------------------------------------------
// Prologue: convert weights to f16 in workspace.
//   W0h   : 256 x 192
//   W1h   : 256 x 256
//   Wouth : last 64 rows of W_out (rows 128..191), 64 x 256
// ---------------------------------------------------------------------------
__global__ void convert_weights(const float* __restrict__ W0,
                                const float* __restrict__ W1,
                                const float* __restrict__ Wout,
                                _Float16* __restrict__ W0h,
                                _Float16* __restrict__ W1h,
                                _Float16* __restrict__ Wouth) {
    int i = blockIdx.x * blockDim.x + threadIdx.x;
    const int n0 = HID * IN_DIM;     // 49152
    const int n1 = HID * HID;        // 65536
    const int n2 = COORD_DIM * HID;  // 16384
    if (i < n0) {
        W0h[i] = (_Float16)W0[i];
    } else if (i < n0 + n1) {
        W1h[i - n0] = (_Float16)W1[i - n0];
    } else if (i < n0 + n1 + n2) {
        int j = i - n0 - n1;
        Wouth[j] = (_Float16)Wout[SPIKE_DIM * HID + j];   // rows 128..191
    }
}

// ---------------------------------------------------------------------------
// Main persistent SNN kernel: one block = 32 batch rows, 16 waves.
// Wave w owns hidden column block [w*16, w*16+16) for both 16-row blocks.
// W1 slice lives in registers (8 x v16h) for all 25 steps.
// ---------------------------------------------------------------------------
__global__ __launch_bounds__(512) void snn_kernel(
    const float* __restrict__ spikes_in, const float* __restrict__ coords_in,
    const float* __restrict__ b0, const float* __restrict__ b1,
    const float* __restrict__ bout,
    const _Float16* __restrict__ W0h, const _Float16* __restrict__ W1h,
    const _Float16* __restrict__ Wouth,
    float* __restrict__ out_action, float* __restrict__ out_coords) {

    __shared__ _Float16 sX[BM][XPAD];   // staged input x (f16), cols 0..191
    __shared__ _Float16 sS[BM][SPAD];   // spike exchange buffer (f16)

    const int tid = threadIdx.x;
    const int w   = tid >> 5;     // wave 0..15
    const int l   = tid & 31;     // lane
    const int ln  = l & 15;       // N column within tile (C/D layout)
    const int lh  = l >> 4;       // half-wave select
    const long m0 = (long)blockIdx.x * BM;

    // ---- Phase 1: stage x = concat(spikes, coords) into LDS as f16 ----
    for (int idx = tid; idx < BM * IN_DIM; idx += 512) {   // 12 uniform iters
        int r = idx / IN_DIM, c = idx - r * IN_DIM;
        float v = (c < SPIKE_DIM)
                      ? spikes_in[(m0 + r) * SPIKE_DIM + c]
                      : coords_in[(m0 + r) * COORD_DIM + (c - SPIKE_DIM)];
        sX[r][c] = (_Float16)v;
    }
    __syncthreads();

    const int ncol = w * 16 + ln;       // this lane's hidden column
    const float b0v = b0[ncol];
    const float b1v = b1[ncol];

    // ---- Phase 2: c0 = x @ W0^T + b0 (kept in accumulator registers) ----
    v8f c0acc[2];
    #pragma unroll
    for (int r = 0; r < 2; ++r) {
        v8f acc;
        #pragma unroll
        for (int j = 0; j < 8; ++j) acc[j] = b0v;
        #pragma unroll
        for (int kc = 0; kc < IN_DIM / 32; ++kc) {
            v16h a, bf;
            const _Float16* ap = &sX[r * 16 + ln][kc * 32 + lh * 8];
            ((v8h*)&a)[0] = *(const v8h*)ap;
            ((v8h*)&a)[1] = *(const v8h*)(ap + 16);
            const _Float16* bp = &W0h[(long)ncol * IN_DIM + kc * 32 + lh * 8];
            ((v8h*)&bf)[0] = *(const v8h*)bp;
            ((v8h*)&bf)[1] = *(const v8h*)(bp + 16);
            acc = __builtin_amdgcn_wmma_f32_16x16x32_f16(
                false, a, false, bf, (short)0, acc, false, false);
        }
        c0acc[r] = acc;
    }

    // ---- Hold this wave's W1 column slice (16 cols x 256 K) in VGPRs ----
    v16h B1[8];
    #pragma unroll
    for (int kc = 0; kc < 8; ++kc) {
        const _Float16* bp = &W1h[(long)ncol * HID + kc * 32 + lh * 8];
        ((v8h*)&B1[kc])[0] = *(const v8h*)bp;
        ((v8h*)&B1[kc])[1] = *(const v8h*)(bp + 16);
    }

    // ---- Membrane state in registers ----
    v8f v0[2], v1[2], s1[2];
    #pragma unroll
    for (int r = 0; r < 2; ++r)
        #pragma unroll
        for (int j = 0; j < 8; ++j) { v0[r][j] = 0.f; v1[r][j] = 0.f; s1[r][j] = 0.f; }

    // ---- Phase 3: 25 recurrent steps ----
    for (int t = 0; t < NUM_STEPS; ++t) {
        // layer-0 LIF -> s0 spikes into LDS (f16)
        #pragma unroll
        for (int r = 0; r < 2; ++r) {
            #pragma unroll
            for (int j = 0; j < 8; ++j) {
                float v  = 0.7f * v0[r][j] + 0.3f * c0acc[r][j];
                float sp = (v >= 1.0f) ? 1.0f : 0.0f;
                v0[r][j] = (sp != 0.0f) ? 0.0f : v;
                sS[r * 16 + lh * 8 + j][ncol] = (_Float16)sp;
            }
        }
        __syncthreads();   // s0 visible to all waves

        // c1 = s0 @ W1^T + b1, then layer-1 LIF
        #pragma unroll
        for (int r = 0; r < 2; ++r) {
            v8f acc;
            #pragma unroll
            for (int j = 0; j < 8; ++j) acc[j] = b1v;
            #pragma unroll
            for (int kc = 0; kc < 8; ++kc) {
                v16h a;
                const _Float16* ap = &sS[r * 16 + ln][kc * 32 + lh * 8];
                ((v8h*)&a)[0] = *(const v8h*)ap;
                ((v8h*)&a)[1] = *(const v8h*)(ap + 16);
                acc = __builtin_amdgcn_wmma_f32_16x16x32_f16(
                    false, a, false, B1[kc], (short)0, acc, false, false);
            }
            #pragma unroll
            for (int j = 0; j < 8; ++j) {
                float v  = 0.7f * v1[r][j] + 0.3f * acc[j];
                float sp = (v >= 1.0f) ? 1.0f : 0.0f;
                v1[r][j] = (sp != 0.0f) ? 0.0f : v;
                s1[r][j] = sp;
            }
        }
        __syncthreads();   // all reads of s0 done before next-step overwrite
    }

    // ---- Phase 4: next_action = s1 (sign of {0,1} spikes), stage s1 to LDS ----
    #pragma unroll
    for (int r = 0; r < 2; ++r) {
        #pragma unroll
        for (int j = 0; j < 8; ++j) {
            long row = m0 + r * 16 + lh * 8 + j;
            out_action[row * HID + ncol] = s1[r][j];
            sS[r * 16 + lh * 8 + j][ncol] = (_Float16)s1[r][j];
        }
    }
    __syncthreads();

    // ---- Phase 5: next_coords = (s1 @ Wout_last64^T + bout_last64) ----
    if (w < 8) {                        // 8 tiles: 2 row-blocks x 4 col-blocks
        const int r  = w & 1;
        const int c  = w >> 1;
        const int oc = c * 16 + ln;     // output coord column 0..63
        float bo = bout[SPIKE_DIM + oc];
        v8f acc;
        #pragma unroll
        for (int j = 0; j < 8; ++j) acc[j] = bo;
        #pragma unroll
        for (int kc = 0; kc < 8; ++kc) {
            v16h a, bf;
            const _Float16* ap = &sS[r * 16 + ln][kc * 32 + lh * 8];
            ((v8h*)&a)[0] = *(const v8h*)ap;
            ((v8h*)&a)[1] = *(const v8h*)(ap + 16);
            const _Float16* bp = &Wouth[(long)oc * HID + kc * 32 + lh * 8];
            ((v8h*)&bf)[0] = *(const v8h*)bp;
            ((v8h*)&bf)[1] = *(const v8h*)(bp + 16);
            acc = __builtin_amdgcn_wmma_f32_16x16x32_f16(
                false, a, false, bf, (short)0, acc, false, false);
        }
        #pragma unroll
        for (int j = 0; j < 8; ++j) {
            long row = m0 + r * 16 + lh * 8 + j;
            out_coords[row * COORD_DIM + oc] = acc[j];
        }
    }
}

// ---------------------------------------------------------------------------
extern "C" void kernel_launch(void* const* d_in, const int* in_sizes, int n_in,
                              void* d_out, int out_size, void* d_ws, size_t ws_size,
                              hipStream_t stream) {
    const float* spikes = (const float*)d_in[0];
    const float* coords = (const float*)d_in[1];
    const float* W0     = (const float*)d_in[2];
    const float* b0     = (const float*)d_in[3];
    const float* W1     = (const float*)d_in[4];
    const float* b1     = (const float*)d_in[5];
    const float* Wout   = (const float*)d_in[6];
    const float* bout   = (const float*)d_in[7];

    const int batch = in_sizes[0] / SPIKE_DIM;

    _Float16* W0h   = (_Float16*)d_ws;                 // 256*192 halves
    _Float16* W1h   = W0h + HID * IN_DIM;              // 256*256 halves
    _Float16* Wouth = W1h + HID * HID;                 // 64*256 halves

    const int nconv = HID * IN_DIM + HID * HID + COORD_DIM * HID;  // 131072
    convert_weights<<<(nconv + 255) / 256, 256, 0, stream>>>(
        W0, W1, Wout, W0h, W1h, Wouth);

    float* out_action = (float*)d_out;                 // (batch, 256)
    float* out_coords = out_action + (long)batch * HID;// (batch, 64)

    snn_kernel<<<batch / BM, 512, 0, stream>>>(
        spikes, coords, b0, b1, bout, W0h, W1h, Wouth, out_action, out_coords);
}
